// BahdanauAttention_53025666236997
// MI455X (gfx1250) — compile-verified, matrix operand order checked by reading, not run on hardware
//
#include <hip/hip_runtime.h>
#include <hip/hip_bf16.h>
#include <stdint.h>

// Bahdanau attention for MI455X (gfx1250).
// B=32, T=2048, FEAT=1024, UNITS=512, HID=512.
// img (256MB fp32) read twice => ~22us HBM floor at 23.3 TB/s.
// img@W1 (68.7 GFLOP) via v_wmma_f32_16x16x32_bf16 with 32x64 register tiles
// per wave, double-buffered async W1^T panels, fused v_tanh_f32 + V-dot epilogue.

#define BATCH  32
#define TLEN   2048
#define FEATN  1024
#define UNITSN 512
#define HIDN   512

#define TOK_TILE   32    // tokens per workgroup (all 4 waves share them)
#define WG_THREADS 128   // 4 wave32s; waves split the unit (N) dimension
#define KCHUNK     64    // K elements staged per async B panel
#define NPANEL     256   // units per pass (4 waves x 64)

typedef __attribute__((ext_vector_type(16))) __bf16 v16bf;
typedef __attribute__((ext_vector_type(8)))  __bf16 v8bf;
typedef __attribute__((ext_vector_type(4)))  __bf16 v4bf;
typedef __attribute__((ext_vector_type(8)))  float  v8f;

// ---------------------------------------------------------------------------
// Kernel 0: W1 [FEAT][UNITS] fp32 -> W1^T [UNITS][FEAT] bf16 (workspace).
// ---------------------------------------------------------------------------
__global__ __launch_bounds__(256) void k_w1_to_bf16T(const float* __restrict__ W1,
                                                     __bf16* __restrict__ w1t)
{
    const int idx = blockIdx.x * 256 + threadIdx.x;   // over UNITS*FEAT
    const int u = idx >> 10;
    const int k = idx & (FEATN - 1);
    w1t[idx] = (__bf16)W1[k * UNITSN + u];
}

// ---------------------------------------------------------------------------
// Kernel 1: proj_hid[b][u] = hidden[b] @ W2 + b2   (tiny)
// ---------------------------------------------------------------------------
__global__ __launch_bounds__(256) void k_projhid(const float* __restrict__ hidden,
                                                 const float* __restrict__ W2,
                                                 const float* __restrict__ b2,
                                                 float* __restrict__ ph)
{
    const int b = blockIdx.x;
    for (int u = threadIdx.x; u < UNITSN; u += 256) {
        float acc = b2[u];
        for (int k = 0; k < HIDN; ++k)
            acc = fmaf(hidden[b * HIDN + k], W2[k * UNITSN + u], acc);
        ph[b * UNITSN + u] = acc;
    }
}

// ---------------------------------------------------------------------------
// Kernel 2: logits[b][t] = sum_u tanh(img[b,t,:]@W1[:,u] + b1[u] + ph[b,u])*V[u]
//
// Workgroup = 4 waves, 32 tokens. Wave w owns units [pass*256 + w*64, +64).
// Per-wave register tile: 32 tokens x 64 units = 8 x v8f accumulators.
// LDS: A tile [32][1024] bf16 (64KB) + double-buffered W1^T K-chunk panels
// [256 units][64 K] bf16 (2 x 32KB) = 128KB -> 2 WGs / 320KB WGP.
// B panels arrive via global_load_async_to_lds_b128 one stage ahead.
// ---------------------------------------------------------------------------
__global__ __launch_bounds__(WG_THREADS) void k_logits(
    const float*  __restrict__ img,     // [B][T][FEAT] fp32
    const __bf16* __restrict__ w1t,     // [UNITS][FEAT] bf16
    const float*  __restrict__ b1,      // [UNITS]
    const float*  __restrict__ ph,      // [B][UNITS]
    const float*  __restrict__ Vv,      // [UNITS]
    float*        __restrict__ logits)  // [B][T]
{
    __shared__ __bf16 ldsA[TOK_TILE * FEATN];          // 64 KB
    __shared__ __bf16 ldsB[2 * NPANEL * KCHUNK];       // 2 x 32 KB

    const int b    = blockIdx.x;
    const int t0   = blockIdx.y * TOK_TILE;
    const int tid  = threadIdx.x;
    const int lane = tid & 31;
    const int wave = tid >> 5;
    const int col  = lane & 15;   // N column / A row within a 16x16 tile
    const int hi   = lane >> 4;   // K half: {0..7,16..23} vs {8..15,24..31}

    // ---- Stage A: fp32 img tile -> bf16 LDS, row-major [tok][feat]
    {
        const float* gA = img + ((size_t)b * TLEN + t0) * FEATN;
        for (int i = tid * 4; i < TOK_TILE * FEATN; i += WG_THREADS * 4) {
            const float4 f = *(const float4*)(gA + i);
            v4bf p;
            p.x = (__bf16)f.x; p.y = (__bf16)f.y;
            p.z = (__bf16)f.z; p.w = (__bf16)f.w;
            *(v4bf*)(&ldsA[i]) = p;
        }
    }

    const unsigned ldsB_base = (unsigned)(uintptr_t)&ldsB[0];

    // Async-stage one W1^T K-chunk panel: 256 units x 64 K bf16 = 32 KB.
    // 2048 x 16B chunks, 16 per thread.
    auto issue_panel = [&](int p, int kc, int buf) {
        const __bf16* gB = w1t + (size_t)p * NPANEL * FEATN + kc * KCHUNK;
        #pragma unroll
        for (int i = 0; i < 16; ++i) {
            const int c  = i * WG_THREADS + tid;
            const int u  = c >> 3;          // unit within panel
            const int j  = c & 7;           // 16B chunk within the 128B row
            unsigned lds_off = ldsB_base + (unsigned)(buf * (NPANEL * KCHUNK * 2)
                                                      + u * (KCHUNK * 2) + j * 16);
            const __bf16* ga = gB + (size_t)u * FEATN + j * 8;
            asm volatile("global_load_async_to_lds_b128 %0, %1, off"
                         :: "v"(lds_off), "v"(ga) : "memory");
        }
    };

    float lp[16];                  // logit partials: [mt 0..1][r 0..7]
    #pragma unroll
    for (int i = 0; i < 16; ++i) lp[i] = 0.f;

    v8f acc[8];                    // [mt][ntt] -> acc[mt*4+ntt]

    const __bf16* Ab = &ldsA[(size_t)col * FEATN + hi * 8];
    const int bElem  = (wave * 64 + col) * KCHUNK + hi * 8;   // within a buffer

    issue_panel(0, 0, 0);

    // 2 unit-passes x 16 K-chunks = 32 double-buffered stages
    #pragma unroll 1
    for (int s = 0; s < 32; ++s) {
        const int p   = s >> 4;
        const int kc  = s & 15;
        const int cur = s & 1;

        asm volatile("s_wait_asynccnt 0" ::: "memory");
        __syncthreads();                       // panel ready; prev stage fully read
        if (s < 31) issue_panel((s + 1) >> 4, (s + 1) & 15, (s + 1) & 1);

        if (kc == 0) {
            #pragma unroll
            for (int i = 0; i < 8; ++i)
                acc[i] = (v8f){0.f,0.f,0.f,0.f,0.f,0.f,0.f,0.f};
        }

        // ---- 2 K-steps of 32: 2 A-frags + 4 B-frags -> 8 WMMAs each
        #pragma unroll
        for (int ks = 0; ks < 2; ++ks) {
            const int k = kc * KCHUNK + ks * 32;
            v16bf a[2], bm[4];
            #pragma unroll
            for (int mt = 0; mt < 2; ++mt) {
                const __bf16* pa = Ab + (size_t)mt * 16 * FEATN + k;
                v8bf lo = *(const v8bf*)pa;
                v8bf hh = *(const v8bf*)(pa + 16);
                a[mt] = __builtin_shufflevector(lo, hh, 0,1,2,3,4,5,6,7,
                                                        8,9,10,11,12,13,14,15);
            }
            #pragma unroll
            for (int nt = 0; nt < 4; ++nt) {
                const __bf16* pb = &ldsB[(size_t)cur * NPANEL * KCHUNK]
                                   + bElem + nt * 16 * KCHUNK + ks * 32;
                v8bf lo = *(const v8bf*)pb;
                v8bf hh = *(const v8bf*)(pb + 16);
                bm[nt] = __builtin_shufflevector(lo, hh, 0,1,2,3,4,5,6,7,
                                                         8,9,10,11,12,13,14,15);
            }
            #pragma unroll
            for (int mt = 0; mt < 2; ++mt)
                #pragma unroll
                for (int nt = 0; nt < 4; ++nt)
                    acc[mt * 4 + nt] = __builtin_amdgcn_wmma_f32_16x16x32_bf16(
                        false, a[mt], false, bm[nt], (short)0,
                        acc[mt * 4 + nt], false, false);
        }

        // ---- end of a unit-pass: fused bias + v_tanh_f32 + dot(V)
        if (kc == 15) {
            #pragma unroll
            for (int nt = 0; nt < 4; ++nt) {
                const int n = p * NPANEL + wave * 64 + nt * 16 + col;
                const float bias = b1[n] + ph[b * UNITSN + n];
                const float vw   = Vv[n];
                #pragma unroll
                for (int mt = 0; mt < 2; ++mt) {
                    #pragma unroll
                    for (int r = 0; r < 8; ++r) {
                        float x = acc[mt * 4 + nt][r] + bias, y;
                        asm("v_tanh_f32 %0, %1\n\tv_nop" : "=v"(y) : "v"(x));
                        lp[mt * 8 + r] += y * vw;
                    }
                }
            }
        }
    }

    // ---- reduce partials over the 16 columns (stays within each lane-half)
    #pragma unroll
    for (int m = 1; m < 16; m <<= 1) {
        #pragma unroll
        for (int i = 0; i < 16; ++i)
            lp[i] += __shfl_xor(lp[i], m, 32);
    }

    // ---- cross-wave reduction (waves hold disjoint unit ranges).
    // ldsA is dead now; alias it after a barrier.
    float* ldsRed = (float*)ldsA;              // [4][32]
    __syncthreads();
    if (col == 0) {
        #pragma unroll
        for (int mt = 0; mt < 2; ++mt)
            #pragma unroll
            for (int r = 0; r < 8; ++r)
                ldsRed[wave * 32 + mt * 16 + hi * 8 + r] = lp[mt * 8 + r];
    }
    __syncthreads();
    if (tid < TOK_TILE) {
        const float s = ldsRed[tid] + ldsRed[32 + tid]
                      + ldsRed[64 + tid] + ldsRed[96 + tid];
        logits[b * TLEN + t0 + tid] = s;
    }
}

// ---------------------------------------------------------------------------
// Kernel 3: masked softmax over T per batch; weights -> d_out.
// (bV is a uniform logit shift -> softmax-invariant.)
// ---------------------------------------------------------------------------
__global__ __launch_bounds__(256) void k_softmax(const float* __restrict__ logits,
                                                 const int*   __restrict__ mask,
                                                 float*       __restrict__ wout)
{
    __shared__ float red[8];
    __shared__ float sstat;
    const int b = blockIdx.x, tid = threadIdx.x;
    const int lane = tid & 31, wv = tid >> 5;

    float vals[8];
    float mx = -3.0e38f;
    #pragma unroll
    for (int i = 0; i < 8; ++i) {
        const int t = tid + i * 256;
        float l = logits[b * TLEN + t];
        if (mask[b * TLEN + t] == 0) l -= 1.0e9f;
        vals[i] = l;
        mx = fmaxf(mx, l);
    }
    #pragma unroll
    for (int m = 16; m >= 1; m >>= 1) mx = fmaxf(mx, __shfl_xor(mx, m, 32));
    if (lane == 0) red[wv] = mx;
    __syncthreads();
    if (tid == 0) {
        float m2 = red[0];
        for (int i = 1; i < 8; ++i) m2 = fmaxf(m2, red[i]);
        sstat = m2;
    }
    __syncthreads();
    mx = sstat;

    float s = 0.f;
    #pragma unroll
    for (int i = 0; i < 8; ++i) { vals[i] = __expf(vals[i] - mx); s += vals[i]; }
    #pragma unroll
    for (int m = 16; m >= 1; m >>= 1) s += __shfl_xor(s, m, 32);
    if (lane == 0) red[wv] = s;
    __syncthreads();
    if (tid == 0) {
        float t2 = 0.f;
        for (int i = 0; i < 8; ++i) t2 += red[i];
        sstat = t2;
    }
    __syncthreads();
    const float inv = 1.0f / sstat;
    #pragma unroll
    for (int i = 0; i < 8; ++i)
        wout[b * TLEN + tid + i * 256] = vals[i] * inv;
}

// ---------------------------------------------------------------------------
// Kernels 4+5: context[b][f] = sum_t w[b][t] * img[b][t][f]
// Second coalesced float4 pass over img (1024 blocks), then 32-way reduce.
// ---------------------------------------------------------------------------
__global__ __launch_bounds__(256) void k_ctx_partial(const float* __restrict__ img,
                                                     const float* __restrict__ w,
                                                     float*       __restrict__ partial)
{
    const int b = blockIdx.x, ts = blockIdx.y, tid = threadIdx.x;
    const int f = tid * 4;
    float4 acc = make_float4(0.f, 0.f, 0.f, 0.f);
    #pragma unroll 4
    for (int j = 0; j < TLEN / 32; ++j) {
        const int t = ts * (TLEN / 32) + j;
        const float wt = w[b * TLEN + t];
        const float4 iv = *(const float4*)(img + ((size_t)b * TLEN + t) * FEATN + f);
        acc.x = fmaf(wt, iv.x, acc.x);
        acc.y = fmaf(wt, iv.y, acc.y);
        acc.z = fmaf(wt, iv.z, acc.z);
        acc.w = fmaf(wt, iv.w, acc.w);
    }
    *(float4*)(partial + ((size_t)(b * 32 + ts)) * FEATN + f) = acc;
}

__global__ __launch_bounds__(256) void k_ctx_reduce(const float* __restrict__ partial,
                                                    float*       __restrict__ ctx)
{
    const int b = blockIdx.x, tid = threadIdx.x;
    const int f = tid * 4;
    float4 acc = make_float4(0.f, 0.f, 0.f, 0.f);
    for (int ts = 0; ts < 32; ++ts) {
        const float4 p = *(const float4*)(partial + ((size_t)(b * 32 + ts)) * FEATN + f);
        acc.x += p.x; acc.y += p.y; acc.z += p.z; acc.w += p.w;
    }
    *(float4*)(ctx + (size_t)b * FEATN + f) = acc;
}

// ---------------------------------------------------------------------------
extern "C" void kernel_launch(void* const* d_in, const int* in_sizes, int n_in,
                              void* d_out, int out_size, void* d_ws, size_t ws_size,
                              hipStream_t stream) {
    (void)in_sizes; (void)n_in; (void)out_size; (void)ws_size;

    const float* img    = (const float*)d_in[0];   // [32][2048][1024]
    const int*   mask   = (const int*)  d_in[1];   // [32][2048]
    const float* hidden = (const float*)d_in[2];   // [32][512]
    const float* W1     = (const float*)d_in[3];   // [1024][512]
    const float* b1     = (const float*)d_in[4];   // [512]
    const float* W2     = (const float*)d_in[5];   // [512][512]
    const float* b2     = (const float*)d_in[6];   // [512]
    const float* Vv     = (const float*)d_in[7];   // [512]
    // d_in[8] = bV: softmax-invariant, unused.

    float* out     = (float*)d_out;
    float* ctx_out = out;                          // [32*1024] context first
    float* w_out   = out + BATCH * FEATN;          // [32*2048] weights second

    char* ws = (char*)d_ws;
    __bf16* w1t    = (__bf16*)ws;                                        // 1 MB
    float*  ph     = (float*)(ws + (1 << 20));                           // 64 KB
    float*  logits = (float*)(ws + (1 << 20) + (64 << 10));              // 256 KB
    float*  part   = (float*)(ws + (1 << 20) + (64 << 10) + (256 << 10));// 4 MB

    k_w1_to_bf16T<<<(UNITSN * FEATN) / 256, 256, 0, stream>>>(W1, w1t);
    k_projhid<<<BATCH, 256, 0, stream>>>(hidden, W2, b2, ph);

    dim3 g2(BATCH, TLEN / TOK_TILE);               // 32 x 64 workgroups
    k_logits<<<g2, WG_THREADS, 0, stream>>>(img, w1t, b1, ph, Vv, logits);

    k_softmax<<<BATCH, 256, 0, stream>>>(logits, mask, w_out);

    dim3 g4(BATCH, 32);
    k_ctx_partial<<<g4, 256, 0, stream>>>(img, w_out, part);
    k_ctx_reduce<<<BATCH, 256, 0, stream>>>(part, ctx_out);
}
